// MoE_23682449670362
// MI455X (gfx1250) — compile-verified, matrix operand order checked by reading, not run on hardware
//
#include <hip/hip_runtime.h>

#define TOPK 2
#define NEXP 8
#define DIM 1024
#define HIDDEN 2816
#define TOKENS 2048            // BS*SLEN
#define ROWS (TOKENS * TOPK)   // 4096 routed rows

typedef __attribute__((ext_vector_type(16))) __bf16 v16bf;
typedef __attribute__((ext_vector_type(8)))  float  v8f;

union AFrag { unsigned int u[8]; v16bf v; };
union BFrag { uint4 q[2]; v16bf v; };

__device__ __forceinline__ unsigned short f32_to_bf16(float f) {
    unsigned int u = __float_as_uint(f);
    u += 0x7FFFu + ((u >> 16) & 1u);   // round-to-nearest-even
    return (unsigned short)(u >> 16);
}

// K index held by (lane, dword i) of a 16-wide bf16 WMMA fragment (16x16x32 layout)
__device__ __forceinline__ int frag_kbase(int half, int i) {
    return (i < 4) ? (half * 8 + 2 * i) : (16 + half * 8 + 2 * (i - 4));
}

// CDNA5 async copy: 16B global -> LDS, tracked by ASYNCcnt (not LOADcnt).
__device__ __forceinline__ void async_cp16(unsigned lds_off, const void* gaddr) {
    asm volatile("global_load_async_to_lds_b128 %0, %1, off"
                 :: "v"(lds_off), "v"((unsigned long long)(uintptr_t)gaddr)
                 : "memory");
}
__device__ __forceinline__ void wait_async0() {
    asm volatile("s_wait_asynccnt 0x0" ::: "memory");
}

// ---------------- 1. Router: softmax(x @ gate_w^T), biased top-2 ----------------
__global__ __launch_bounds__(256)
void router_kernel(const float* __restrict__ x, const float* __restrict__ gw,
                   const float* __restrict__ bias, int* __restrict__ sel,
                   float* __restrict__ topsc) {
    const int t = blockIdx.x;
    const int wave = threadIdx.x >> 5, lane = threadIdx.x & 31;
    __shared__ float logits[NEXP];
    const float* xr = x + (size_t)t * DIM;
    const float* gr = gw + (size_t)wave * DIM;
    float acc = 0.f;
    for (int d = lane; d < DIM; d += 32) acc += xr[d] * gr[d];
#pragma unroll
    for (int off = 16; off > 0; off >>= 1) acc += __shfl_xor(acc, off, 32);
    if (lane == 0) logits[wave] = acc;
    __syncthreads();
    if (threadIdx.x == 0) {
        float p[NEXP];
        float mx = logits[0];
        for (int e = 1; e < NEXP; ++e) mx = fmaxf(mx, logits[e]);
        float s = 0.f;
        for (int e = 0; e < NEXP; ++e) { p[e] = __expf(logits[e] - mx); s += p[e]; }
        float inv = 1.f / s;
        for (int e = 0; e < NEXP; ++e) p[e] *= inv;
        int s0 = 0; float b0 = -3.4e38f;
        for (int e = 0; e < NEXP; ++e) { float v = p[e] + bias[e]; if (v > b0) { b0 = v; s0 = e; } }
        int s1 = 0; float b1 = -3.4e38f;
        for (int e = 0; e < NEXP; ++e) {
            if (e == s0) continue;
            float v = p[e] + bias[e]; if (v > b1) { b1 = v; s1 = e; }
        }
        sel[2 * t] = s0; sel[2 * t + 1] = s1;
        topsc[2 * t] = p[s0]; topsc[2 * t + 1] = p[s1];
    }
}

// ---------------- 2. Deterministic stable counting sort (no atomics) ----------------
__global__ void rank_kernel(const int* __restrict__ sel, const float* __restrict__ topsc,
                            int* __restrict__ pos, int* __restrict__ tok_of_p,
                            float* __restrict__ sc_of_p, int* __restrict__ seg) {
    __shared__ int cnt[NEXP], st[NEXP];
    const int tid = threadIdx.x;
    if (tid < NEXP) {
        int c = 0;
        for (int i = 0; i < ROWS; ++i) c += (sel[i] == tid);
        cnt[tid] = c;
    }
    __syncthreads();
    if (tid == 0) {
        int a = 0;
        for (int e = 0; e < NEXP; ++e) { st[e] = a; a += cnt[e]; }
    }
    __syncthreads();
    if (tid < NEXP) {
        seg[tid] = st[tid];
        seg[NEXP + tid] = st[tid] + cnt[tid];
        int cur = st[tid];
        for (int i = 0; i < ROWS; ++i) {
            if (sel[i] == tid) {
                pos[i] = cur;
                tok_of_p[cur] = i >> 1;
                sc_of_p[cur] = topsc[i];
                ++cur;
            }
        }
    }
}

// ---------------- 3. Dispatch: gather + scale + fp32->bf16 ----------------
__global__ __launch_bounds__(256)
void dispatch_kernel(const float* __restrict__ x, const int* __restrict__ tok_of_p,
                     const float* __restrict__ sc_of_p, unsigned short* __restrict__ routed) {
    const int idx = blockIdx.x * 256 + threadIdx.x;
    const int p = idx / DIM, d = idx - p * DIM;
    routed[idx] = f32_to_bf16(x[(size_t)tok_of_p[p] * DIM + d] * sc_of_p[p]);
}

// ---------------- 4. Weight swizzle: fp32 [E][K][N] -> bf16 WMMA B-fragment order ----------------
// frag(e, nt, kt): 512 contiguous bf16, lane L owns bytes [L*32, L*32+32)
__global__ __launch_bounds__(256)
void swizzle_kernel(const float* __restrict__ src, unsigned short* __restrict__ dst,
                    int K, int N) {
    const int f = blockIdx.x * 256 + threadIdx.x;
    const int j = f & 15;
    const int L = (f >> 4) & 31;
    const int KT = K >> 5, NT = N >> 4;
    int r = f >> 9;
    const int kt = r % KT; r /= KT;
    const int nt = r % NT; const int e = r / NT;
    const int half = L >> 4;
    const int k = kt * 32 + ((j < 8) ? (half * 8 + j) : (16 + half * 8 + (j - 8)));
    const int n = nt * 16 + (L & 15);
    dst[f] = f32_to_bf16(src[(size_t)e * K * N + (size_t)k * N + n]);
}

// ---------------- 5. Fused GEMM1: H = silu(routed@w1) * (routed@w3), bf16 out ----------------
// BM=128 (8 waves x 16 rows), BN=64. B frags (8KB/K-step) async-staged to LDS, double buffered.
__global__ __launch_bounds__(256, 1)
void gemm1_kernel(const unsigned short* __restrict__ A,   // routed bf16 [ROWS][DIM]
                  const unsigned short* __restrict__ B1,  // w1 frags
                  const unsigned short* __restrict__ B3,  // w3 frags
                  const int* __restrict__ seg,
                  unsigned short* __restrict__ H) {       // [ROWS][HIDDEN] bf16
    const int e = blockIdx.z;
    const int end = seg[NEXP + e];
    const int m0 = seg[e] + blockIdx.y * 128;
    if (m0 >= end) return;
    const int wave = threadIdx.x >> 5, lane = threadIdx.x & 31;
    const int half = lane >> 4, mr = lane & 15;
    const int rowb = m0 + wave * 16;
    const int n0 = blockIdx.x * 64;
    const int KT = DIM / 32;

    __shared__ alignas(16) unsigned short lds[2][8 * 512];  // [buf][mat*4+nt frag][512]

    // async-stage one K-step of B1/B3 fragments (8 frags x 1KB = 512 uint4)
    auto stage = [&](int buf, int kt) {
#pragma unroll
        for (int w = 0; w < 2; ++w) {
            const int idx = threadIdx.x * 2 + w;        // 0..511
            const int frag = idx >> 6, within = idx & 63;
            const unsigned short* src = ((frag < 4) ? B1 : B3) +
                (((long)e * (HIDDEN / 16) + (n0 >> 4) + (frag & 3)) * (long)KT + kt) * 512;
            async_cp16((unsigned)(uintptr_t)&lds[buf][frag * 512 + within * 8],
                       (const uint4*)src + within);
        }
    };
    auto load_a = [&](AFrag& a, int kt, int arow) {
#pragma unroll
        for (int i = 0; i < 8; ++i)
            a.u[i] = *(const unsigned int*)(A + (size_t)arow * DIM + kt * 32 + frag_kbase(half, i));
    };

    v8f acc1[4] = {v8f{}, v8f{}, v8f{}, v8f{}};
    v8f acc3[4] = {v8f{}, v8f{}, v8f{}, v8f{}};
    int arow = rowb + mr; if (arow >= end) arow = end - 1;  // clamp: keep EXEC all-1s

    stage(0, 0);
    AFrag a_cur; load_a(a_cur, 0, arow);

    int buf = 0;
    for (int kt = 0; kt < KT; ++kt) {
        wait_async0();
        __syncthreads();                       // staged buffer visible; prev ds reads done
        if (kt + 1 < KT) stage(buf ^ 1, kt + 1);
        AFrag a_nxt;
        load_a(a_nxt, (kt + 1 < KT) ? kt + 1 : kt, arow);

        // Issue ALL fragment loads, then fence the scheduler so they cannot be
        // sunk to their uses: the dscnt waits become a descending ladder and
        // only the first WMMA pays LDS latency.
        BFrag b1[4], b3[4];
#pragma unroll
        for (int nt = 0; nt < 4; ++nt) {
            const uint4* l1 = (const uint4*)&lds[buf][nt * 512];
            const uint4* l3 = (const uint4*)&lds[buf][(4 + nt) * 512];
            b1[nt].q[0] = l1[lane * 2]; b1[nt].q[1] = l1[lane * 2 + 1];
            b3[nt].q[0] = l3[lane * 2]; b3[nt].q[1] = l3[lane * 2 + 1];
        }
        __builtin_amdgcn_sched_barrier(0);
#pragma unroll
        for (int nt = 0; nt < 4; ++nt) {
            acc1[nt] = __builtin_amdgcn_wmma_f32_16x16x32_bf16(
                false, a_cur.v, false, b1[nt].v, (short)0, acc1[nt], false, false);
            acc3[nt] = __builtin_amdgcn_wmma_f32_16x16x32_bf16(
                false, a_cur.v, false, b3[nt].v, (short)0, acc3[nt], false, false);
        }
        a_cur = a_nxt;
        buf ^= 1;
    }
#pragma unroll
    for (int nt = 0; nt < 4; ++nt) {
#pragma unroll
        for (int r = 0; r < 8; ++r) {
            const int row = rowb + r + 8 * half;
            if (row < end) {
                const float g = acc1[nt][r];
                const float h = (g / (1.f + __expf(-g))) * acc3[nt][r];
                H[(size_t)row * HIDDEN + n0 + nt * 16 + mr] = f32_to_bf16(h);
            }
        }
    }
}

// ---------------- 6. GEMM2: routed_out = H @ w2, fp32 out ----------------
__global__ __launch_bounds__(256, 1)
void gemm2_kernel(const unsigned short* __restrict__ A,   // H bf16 [ROWS][HIDDEN]
                  const unsigned short* __restrict__ B2,  // w2 frags
                  const int* __restrict__ seg,
                  float* __restrict__ O) {                 // [ROWS][DIM] f32
    const int e = blockIdx.z;
    const int end = seg[NEXP + e];
    const int m0 = seg[e] + blockIdx.y * 128;
    if (m0 >= end) return;
    const int wave = threadIdx.x >> 5, lane = threadIdx.x & 31;
    const int half = lane >> 4, mr = lane & 15;
    const int rowb = m0 + wave * 16;
    const int n0 = blockIdx.x * 64;
    const int KT = HIDDEN / 32;

    __shared__ alignas(16) unsigned short lds[2][4 * 512];  // [buf][nt frag][512]

    auto stage = [&](int buf, int kt) {
        const int idx = threadIdx.x;                 // 0..255 (4 frags x 64 uint4)
        const int frag = idx >> 6, within = idx & 63;
        const unsigned short* src = B2 +
            (((long)e * (DIM / 16) + (n0 >> 4) + frag) * (long)KT + kt) * 512;
        async_cp16((unsigned)(uintptr_t)&lds[buf][frag * 512 + within * 8],
                   (const uint4*)src + within);
    };
    auto load_a = [&](AFrag& a, int kt, int arow) {
#pragma unroll
        for (int i = 0; i < 8; ++i)
            a.u[i] = *(const unsigned int*)(A + (size_t)arow * HIDDEN + kt * 32 + frag_kbase(half, i));
    };

    v8f acc[4] = {v8f{}, v8f{}, v8f{}, v8f{}};
    int arow = rowb + mr; if (arow >= end) arow = end - 1;

    stage(0, 0);
    AFrag a_cur; load_a(a_cur, 0, arow);

    int buf = 0;
    for (int kt = 0; kt < KT; ++kt) {
        wait_async0();
        __syncthreads();
        if (kt + 1 < KT) stage(buf ^ 1, kt + 1);
        AFrag a_nxt;
        load_a(a_nxt, (kt + 1 < KT) ? kt + 1 : kt, arow);

        BFrag b[4];
#pragma unroll
        for (int nt = 0; nt < 4; ++nt) {
            const uint4* lp = (const uint4*)&lds[buf][nt * 512];
            b[nt].q[0] = lp[lane * 2]; b[nt].q[1] = lp[lane * 2 + 1];
        }
        __builtin_amdgcn_sched_barrier(0);
#pragma unroll
        for (int nt = 0; nt < 4; ++nt) {
            acc[nt] = __builtin_amdgcn_wmma_f32_16x16x32_bf16(
                false, a_cur.v, false, b[nt].v, (short)0, acc[nt], false, false);
        }
        a_cur = a_nxt;
        buf ^= 1;
    }
#pragma unroll
    for (int nt = 0; nt < 4; ++nt) {
#pragma unroll
        for (int r = 0; r < 8; ++r) {
            const int row = rowb + r + 8 * half;
            if (row < end)
                O[(size_t)row * DIM + n0 + nt * 16 + mr] = acc[nt][r];
        }
    }
}

// ---------------- 7. Combine: out[t] = ro[pos[2t]] + ro[pos[2t+1]] ----------------
__global__ __launch_bounds__(256)
void combine_kernel(const float* __restrict__ ro, const int* __restrict__ pos,
                    float* __restrict__ out) {
    const int idx = blockIdx.x * 256 + threadIdx.x;
    const int t = idx / DIM, d = idx - t * DIM;
    out[idx] = ro[(size_t)pos[2 * t] * DIM + d] + ro[(size_t)pos[2 * t + 1] * DIM + d];
}

extern "C" void kernel_launch(void* const* d_in, const int* in_sizes, int n_in,
                              void* d_out, int out_size, void* d_ws, size_t ws_size,
                              hipStream_t stream) {
    (void)in_sizes; (void)n_in; (void)out_size; (void)ws_size;
    const float* x    = (const float*)d_in[0];
    const float* gw   = (const float*)d_in[1];
    const float* bias = (const float*)d_in[2];
    const float* w1   = (const float*)d_in[3];
    const float* w2   = (const float*)d_in[4];
    const float* w3   = (const float*)d_in[5];
    float* out = (float*)d_out;

    // Workspace carve (1 KB aligned so fragment b128 loads are aligned)
    char* base = (char*)d_ws;
    size_t off = 0;
    auto carve = [&](size_t bytes) { char* p = base + off; off = (off + bytes + 1023) & ~(size_t)1023; return p; };
    unsigned short* w1f    = (unsigned short*)carve((size_t)NEXP * DIM * HIDDEN * 2);
    unsigned short* w3f    = (unsigned short*)carve((size_t)NEXP * DIM * HIDDEN * 2);
    unsigned short* w2f    = (unsigned short*)carve((size_t)NEXP * HIDDEN * DIM * 2);
    unsigned short* routed = (unsigned short*)carve((size_t)ROWS * DIM * 2);
    unsigned short* Hbuf   = (unsigned short*)carve((size_t)ROWS * HIDDEN * 2);
    float* ro        = (float*)carve((size_t)ROWS * DIM * 4);
    float* topsc     = (float*)carve(ROWS * 4);
    float* sc_of_p   = (float*)carve(ROWS * 4);
    int*   sel       = (int*)carve(ROWS * 4);
    int*   pos       = (int*)carve(ROWS * 4);
    int*   tok_of_p  = (int*)carve(ROWS * 4);
    int*   seg       = (int*)carve(2 * NEXP * 4);

    router_kernel<<<TOKENS, 256, 0, stream>>>(x, gw, bias, sel, topsc);
    rank_kernel<<<1, 32, 0, stream>>>(sel, topsc, pos, tok_of_p, sc_of_p, seg);
    dispatch_kernel<<<(ROWS * DIM) / 256, 256, 0, stream>>>(x, tok_of_p, sc_of_p, routed);

    swizzle_kernel<<<(NEXP * DIM * HIDDEN) / 256, 256, 0, stream>>>(w1, w1f, DIM, HIDDEN);
    swizzle_kernel<<<(NEXP * DIM * HIDDEN) / 256, 256, 0, stream>>>(w3, w3f, DIM, HIDDEN);
    swizzle_kernel<<<(NEXP * HIDDEN * DIM) / 256, 256, 0, stream>>>(w2, w2f, HIDDEN, DIM);

    gemm1_kernel<<<dim3(HIDDEN / 64, ROWS / 128, NEXP), 256, 0, stream>>>(routed, w1f, w3f, seg, Hbuf);
    gemm2_kernel<<<dim3(DIM / 64, ROWS / 128, NEXP), 256, 0, stream>>>(Hbuf, w2f, seg, ro);

    combine_kernel<<<(TOKENS * DIM) / 256, 256, 0, stream>>>(ro, pos, out);
}
